// AttentionBlock3D_30416958390774
// MI455X (gfx1250) — matrix operand order repaired and checked by reading, not compiled, not run on hardware
//
#include <hip/hip_runtime.h>

// AttentionBlock3D for MI455X (gfx1250, wave32, WMMA f32_16x16x32_f16)
// B=4, C=256, N=D*H*W=4096, 8 groups.
//
// Layout choices make every LDS staging copy transpose-free so it can use
// GLOBAL_LOAD_ASYNC_TO_LDS_B128:
//   h  : [B][N][C]  (QKV GEMM B-operand, K(=c)-major)
//   q  : [B][N][C]  (QK^T A-operand)
//   k  : [B][N][C]  (QK^T B-operand tile -> LDS [m][c], flat copy)
//   v  : [B][C][N]  (P.V  B-operand tile -> LDS [c][m], row-strided copy)
//   o  : [B][N][C]  (proj GEMM B-operand)

#define B_   4
#define C_   256
#define N_   4096
#define G_   8
#define CG_  32
#define EPSV 1e-5f
#define KT   32   // key tile for flash attention
#define BK   64   // GEMM k-tile (2 WMMA k-steps per stage)

typedef __attribute__((ext_vector_type(16))) _Float16 v16h;
typedef __attribute__((ext_vector_type(8)))  _Float16 v8h;
typedef __attribute__((ext_vector_type(8)))  float    v8f;
typedef __attribute__((ext_vector_type(4)))  float    v4f;
typedef int v4i_ __attribute__((vector_size(4 * sizeof(int))));

// ---------------- async global->LDS copy (16B granule per lane) ------------
#if __has_builtin(__builtin_amdgcn_global_load_async_to_lds_b128)
#define HAS_ASYNC_LDS 1
#endif

__device__ inline void cp16(void* lds, const void* g) {
#ifdef HAS_ASYNC_LDS
  __builtin_amdgcn_global_load_async_to_lds_b128(
      (__attribute__((address_space(1))) v4i_*)(g),
      (__attribute__((address_space(3))) v4i_*)(lds), 0, 0);
#else
  *(v8h*)lds = *(const v8h*)g;
#endif
}
__device__ inline void cp16_wait() {
#ifdef HAS_ASYNC_LDS
  asm volatile("s_wait_asynccnt 0x0" ::: "memory");
#endif
}

// ---- WMMA fragment helpers (CDNA5 wave32 layouts, CDNA5 ISA) ----
// A 16x32 f16: lane L holds row (L&15); halves 0..7 = K (L>>4)*8+0..7,
// halves 8..15 = K 16+(L>>4)*8+0..7  -> two contiguous 16B reads per lane.
__device__ inline v16h load_a_frag(const _Float16* tile, int ld, int lane) {
  const _Float16* p = tile + (lane & 15) * ld + ((lane >> 4) << 3);
  v8h lo = *(const v8h*)p;
  v8h hi = *(const v8h*)(p + 16);
  v16h r;
#pragma unroll
  for (int i = 0; i < 8; ++i) { r[i] = lo[i]; r[i + 8] = hi[i]; }
  return r;
}
// B 32x16 f16 with tile stored [n][k] (K-major per column): lane L holds
// column (L&15), halves 0..7 = K (L>>4)*8+0..7, halves 8..15 = K 16+(L>>4)*8+0..7.
__device__ inline v16h load_b_frag(const _Float16* tile, int ld, int lane) {
  const _Float16* p = tile + (lane & 15) * ld + ((lane >> 4) << 3);
  v8h lo = *(const v8h*)p;
  v8h hi = *(const v8h*)(p + 16);
  v16h r;
#pragma unroll
  for (int i = 0; i < 8; ++i) { r[i] = lo[i]; r[i + 8] = hi[i]; }
  return r;
}
__device__ inline v8f wmma_f16(v16h a, v16h b, v8f c) {
  return __builtin_amdgcn_wmma_f32_16x16x32_f16(false, a, false, b, (short)0, c,
                                                false, false);
}

// ------------------------------- utilities ---------------------------------
__global__ void f32_to_f16_kernel(const float* __restrict__ src,
                                  _Float16* __restrict__ dst, int n) {
  int i = blockIdx.x * blockDim.x + threadIdx.x;
  if (i < n) dst[i] = (_Float16)src[i];
}

// -------- GroupNorm: stats per (batch, group), then normalize to f16 -------
__global__ __launch_bounds__(256) void gn_stats_kernel(
    const float* __restrict__ x, float* __restrict__ stats) {
  const int bg = blockIdx.x;  // 0..31
  const v4f* pv = (const v4f*)(x + (size_t)bg * (CG_ * N_));
  float s = 0.f, s2 = 0.f;
  for (int i = threadIdx.x; i < (CG_ * N_) / 4; i += 256) {
    v4f v = pv[i];
#pragma unroll
    for (int k = 0; k < 4; ++k) { s += v[k]; s2 += v[k] * v[k]; }
  }
  __shared__ float rs[256], rs2[256];
  rs[threadIdx.x] = s; rs2[threadIdx.x] = s2;
  __syncthreads();
  for (int off = 128; off > 0; off >>= 1) {
    if ((int)threadIdx.x < off) {
      rs[threadIdx.x]  += rs[threadIdx.x + off];
      rs2[threadIdx.x] += rs2[threadIdx.x + off];
    }
    __syncthreads();
  }
  if (threadIdx.x == 0) {
    const float inv_n = 1.0f / (float)(CG_ * N_);
    float mu  = rs[0] * inv_n;
    float var = rs2[0] * inv_n - mu * mu;
    stats[2 * bg]     = mu;
    stats[2 * bg + 1] = rsqrtf(var + EPSV);
  }
}

// h stored transposed: h[b][n][c] (8 channels per thread, contiguous write)
__global__ __launch_bounds__(256) void gn_apply_kernel(
    const float* __restrict__ x, const float* __restrict__ gamma,
    const float* __restrict__ beta, const float* __restrict__ stats,
    _Float16* __restrict__ h) {
  int t = blockIdx.x * 256 + threadIdx.x;     // 0 .. B*N*(C/8)-1
  int c0 = (t & 31) << 3;                     // 0..248 step 8 (same group)
  int n  = (t >> 5) & (N_ - 1);
  int b  = t >> 17;                           // / (32*4096)
  int sidx = (b * G_ + (c0 >> 5)) * 2;
  float mu = stats[sidx], rstd = stats[sidx + 1];
  const float* src = x + ((size_t)b * C_ + c0) * N_ + n;  // stride N_
  v8h o;
#pragma unroll
  for (int i = 0; i < 8; ++i)
    o[i] = (_Float16)((src[(size_t)i * N_] - mu) * rstd * gamma[c0 + i] + beta[c0 + i]);
  *(v8h*)(h + ((size_t)b * N_ + n) * C_ + c0) = o;
}

// ------------- QKV GEMM: [768x256] x [256x4096] per batch ------------------
// Block tile 128x128, 8 waves in 2x4 grid, each wave 64x32 (4x2 WMMA tiles).
// A = qkv_w[o][c] (row-major), B = h[b][n][c] (K-major) -> both flat-staged.
__global__ __launch_bounds__(256) void qkv_gemm_kernel(
    const _Float16* __restrict__ W, const _Float16* __restrict__ H,
    const float* __restrict__ bias, _Float16* __restrict__ qb,
    _Float16* __restrict__ kb, _Float16* __restrict__ vb) {
  __shared__ __align__(16) _Float16 As[128][BK];  // [row][k]
  __shared__ __align__(16) _Float16 Bs[128][BK];  // [n][k]
  const int b = blockIdx.z, mblk = blockIdx.y, nblk = blockIdx.x;
  const int tid = threadIdx.x, lane = tid & 31, wave = tid >> 5;
  const int wr = wave & 1, wc = wave >> 1;
  const _Float16* Hb = H + (size_t)b * N_ * C_;
  v8f acc[4][2] = {};
  for (int k0 = 0; k0 < C_; k0 += BK) {
    // stage A and B: 128 rows x 64 halves each = 1024 granules of 16B apiece
#pragma unroll
    for (int i = 0; i < 4; ++i) {
      int f = i * 256 + tid, row = f >> 3, g = (f & 7) << 3;
      cp16(&As[row][g], W  + (size_t)(mblk * 128 + row) * C_ + k0 + g);
      cp16(&Bs[row][g], Hb + (size_t)(nblk * 128 + row) * C_ + k0 + g);
    }
    cp16_wait();
    __syncthreads();
#pragma unroll
    for (int ks = 0; ks < 2; ++ks) {
      v16h bf[2];
#pragma unroll
      for (int j = 0; j < 2; ++j)
        bf[j] = load_b_frag(&Bs[wc * 32 + j * 16][ks * 32], BK, lane);
#pragma unroll
      for (int i = 0; i < 4; ++i) {
        v16h a = load_a_frag(&As[wr * 64 + i * 16][ks * 32], BK, lane);
#pragma unroll
        for (int j = 0; j < 2; ++j) acc[i][j] = wmma_f16(a, bf[j], acc[i][j]);
      }
    }
    __syncthreads();
  }
  const int cl = lane & 15, rh = (lane >> 4) << 3;
  const int sect = (mblk * 128) >> 8;  // 0=q, 1=k, 2=v
#pragma unroll
  for (int i = 0; i < 4; ++i)
#pragma unroll
    for (int j = 0; j < 2; ++j)
#pragma unroll
      for (int r = 0; r < 8; ++r) {
        int o = mblk * 128 + wr * 64 + i * 16 + rh + r;
        int n = nblk * 128 + wc * 32 + j * 16 + cl;
        _Float16 hv = (_Float16)(acc[i][j][r] + bias[o]);
        int c = o & 255;
        if (sect == 2) vb[((size_t)b * C_ + c) * N_ + n] = hv;          // [c][n]
        else {
          _Float16* dst = sect ? kb : qb;                               // [n][c]
          dst[((size_t)b * N_ + n) * C_ + c] = hv;
        }
      }
}

// -------------------- Flash attention, 128 queries / block -----------------
__global__ __launch_bounds__(256) void attn_kernel(
    const _Float16* __restrict__ qb, const _Float16* __restrict__ kb,
    const _Float16* __restrict__ vb, _Float16* __restrict__ ob) {
  __shared__ __align__(16) _Float16 Ks[KT][C_];     // [m][c]  (flat copy)
  __shared__ __align__(16) _Float16 Vs[C_][KT];     // [c][m]  (row-strided copy)
  __shared__ __align__(16) _Float16 Ps[8][16][KT];  // per-wave P staging
  const int b = blockIdx.y, q0 = blockIdx.x * 128;
  const int tid = threadIdx.x, lane = tid & 31, wave = tid >> 5;

  // Q A-fragments for this wave's 16 rows stay in registers (8 x v16h).
  const _Float16* Qrow = qb + ((size_t)b * N_ + q0 + wave * 16) * C_;
  v16h qa[8];
#pragma unroll
  for (int kk = 0; kk < 8; ++kk) qa[kk] = load_a_frag(Qrow + kk * 32, C_, lane);

  v8f Oacc[16] = {};
  float m_i[8], l_i[8];
#pragma unroll
  for (int r = 0; r < 8; ++r) { m_i[r] = -3.0e38f; l_i[r] = 0.f; }
  const float scale = 0.0625f;  // C^-0.5

  for (int j0 = 0; j0 < N_; j0 += KT) {
    __syncthreads();
    {  // K tile: kb rows [j0, j0+KT) are contiguous -> flat 16KB copy
      const _Float16* src = kb + ((size_t)b * N_ + j0) * C_;
#pragma unroll
      for (int i = 0; i < 4; ++i) {
        int f = (i * 256 + tid) << 3;  // half offset, 16B granules
        cp16((_Float16*)Ks + f, src + f);
      }
    }
    {  // V tile: vb[b][c][j0..j0+31] -> Vs[c][0..31], 64B per row
      const _Float16* src = vb + ((size_t)b * C_ + tid) * N_ + j0;
#pragma unroll
      for (int g = 0; g < 4; ++g) cp16(&Vs[tid][g << 3], src + (g << 3));
    }
    cp16_wait();
    __syncthreads();

    // S = Q K^T : 16 rows x 32 keys, K-dim 256
    v8f S[2] = {};
#pragma unroll
    for (int kk = 0; kk < 8; ++kk)
#pragma unroll
      for (int j = 0; j < 2; ++j) {
        v16h bbf = load_b_frag(&Ks[j * 16][kk * 32], C_, lane);
        S[j] = wmma_f16(qa[kk], bbf, S[j]);
      }

    // Online softmax: row r+8*(lane>>4) lives across one 16-lane half.
#pragma unroll
    for (int r = 0; r < 8; ++r) {
      float mx = fmaxf(S[0][r], S[1][r]) * scale;
#pragma unroll
      for (int off = 1; off < 16; off <<= 1) mx = fmaxf(mx, __shfl_xor(mx, off, 32));
      float mnew  = fmaxf(m_i[r], mx);
      float alpha = __expf(m_i[r] - mnew);
      m_i[r] = mnew;
      float rs = 0.f;
#pragma unroll
      for (int j = 0; j < 2; ++j) {
        float e = __expf(S[j][r] * scale - mnew);
        S[j][r] = e;
        rs += e;
      }
#pragma unroll
      for (int off = 1; off < 16; off <<= 1) rs += __shfl_xor(rs, off, 32);
      l_i[r] = l_i[r] * alpha + rs;
#pragma unroll
      for (int t = 0; t < 16; ++t) Oacc[t][r] *= alpha;
    }

    {  // reshape P: D-frag -> A-frag via per-wave LDS patch
      const int cl = lane & 15, rh = (lane >> 4) << 3;
#pragma unroll
      for (int j = 0; j < 2; ++j)
#pragma unroll
        for (int r = 0; r < 8; ++r)
          Ps[wave][rh + r][j * 16 + cl] = (_Float16)S[j][r];
    }
    asm volatile("s_wait_dscnt 0x0" ::: "memory");  // wave-local LDS RAW

    // O += P @ V (1 k-step of 32, 16 column tiles over C=256)
    v16h pa = load_a_frag(&Ps[wave][0][0], KT, lane);
#pragma unroll
    for (int t = 0; t < 16; ++t) {
      v16h bbf = load_b_frag(&Vs[t * 16][0], KT, lane);
      Oacc[t] = wmma_f16(pa, bbf, Oacc[t]);
    }
  }

  // finalize: divide by l, store to ob[b][n][c] (K-major for proj GEMM)
  const int cl = lane & 15, rh = (lane >> 4) << 3;
#pragma unroll
  for (int r = 0; r < 8; ++r) {
    float inv = 1.0f / l_i[r];
    int n = q0 + wave * 16 + rh + r;
#pragma unroll
    for (int t = 0; t < 16; ++t) {
      int c = t * 16 + cl;
      ob[((size_t)b * N_ + n) * C_ + c] = (_Float16)(Oacc[t][r] * inv);
    }
  }
}

// --------- Proj GEMM [256x256]x[256x4096] + bias + residual (f32 out) ------
__global__ __launch_bounds__(256) void proj_gemm_kernel(
    const _Float16* __restrict__ W, const _Float16* __restrict__ O,
    const float* __restrict__ bias, const float* __restrict__ x,
    float* __restrict__ out) {
  __shared__ __align__(16) _Float16 As[128][BK];
  __shared__ __align__(16) _Float16 Bs[128][BK];
  const int b = blockIdx.z, mblk = blockIdx.y, nblk = blockIdx.x;
  const int tid = threadIdx.x, lane = tid & 31, wave = tid >> 5;
  const int wr = wave & 1, wc = wave >> 1;
  const _Float16* Ob = O + (size_t)b * N_ * C_;  // [n][c]
  v8f acc[4][2] = {};
  for (int k0 = 0; k0 < C_; k0 += BK) {
#pragma unroll
    for (int i = 0; i < 4; ++i) {
      int f = i * 256 + tid, row = f >> 3, g = (f & 7) << 3;
      cp16(&As[row][g], W  + (size_t)(mblk * 128 + row) * C_ + k0 + g);
      cp16(&Bs[row][g], Ob + (size_t)(nblk * 128 + row) * C_ + k0 + g);
    }
    cp16_wait();
    __syncthreads();
#pragma unroll
    for (int ks = 0; ks < 2; ++ks) {
      v16h bf[2];
#pragma unroll
      for (int j = 0; j < 2; ++j)
        bf[j] = load_b_frag(&Bs[wc * 32 + j * 16][ks * 32], BK, lane);
#pragma unroll
      for (int i = 0; i < 4; ++i) {
        v16h a = load_a_frag(&As[wr * 64 + i * 16][ks * 32], BK, lane);
#pragma unroll
        for (int j = 0; j < 2; ++j) acc[i][j] = wmma_f16(a, bf[j], acc[i][j]);
      }
    }
    __syncthreads();
  }
  const int cl = lane & 15, rh = (lane >> 4) << 3;
#pragma unroll
  for (int i = 0; i < 4; ++i)
#pragma unroll
    for (int j = 0; j < 2; ++j)
#pragma unroll
      for (int r = 0; r < 8; ++r) {
        int o = mblk * 128 + wr * 64 + i * 16 + rh + r;
        int n = nblk * 128 + wc * 32 + j * 16 + cl;
        size_t idx = ((size_t)b * C_ + o) * N_ + n;
        out[idx] = acc[i][j][r] + bias[o] + x[idx];
      }
}

// --------------------------------- launch ----------------------------------
extern "C" void kernel_launch(void* const* d_in, const int* in_sizes, int n_in,
                              void* d_out, int out_size, void* d_ws, size_t ws_size,
                              hipStream_t stream) {
  const float* x      = (const float*)d_in[0];
  const float* gamma  = (const float*)d_in[1];
  const float* beta   = (const float*)d_in[2];
  const float* qkv_w  = (const float*)d_in[3];
  const float* qkv_b  = (const float*)d_in[4];
  const float* proj_w = (const float*)d_in[5];
  const float* proj_b = (const float*)d_in[6];
  float* out = (float*)d_out;

  char* w = (char*)d_ws;
  float*    stats = (float*)w;    w += 1024;
  _Float16* h     = (_Float16*)w; w += (size_t)B_ * N_ * C_ * 2;  // [b][n][c]
  _Float16* wqkv  = (_Float16*)w; w += (size_t)768 * 256 * 2;
  _Float16* wproj = (_Float16*)w; w += (size_t)256 * 256 * 2;
  _Float16* qb    = (_Float16*)w; w += (size_t)B_ * N_ * C_ * 2;  // [b][n][c]
  _Float16* kb    = (_Float16*)w; w += (size_t)B_ * N_ * C_ * 2;  // [b][n][c]
  _Float16* vb    = (_Float16*)w; w += (size_t)B_ * C_ * N_ * 2;  // [b][c][n]
  _Float16* ob    = (_Float16*)w; w += (size_t)B_ * N_ * C_ * 2;  // [b][n][c]

  f32_to_f16_kernel<<<(768 * 256) / 256, 256, 0, stream>>>(qkv_w, wqkv, 768 * 256);
  f32_to_f16_kernel<<<(256 * 256) / 256, 256, 0, stream>>>(proj_w, wproj, 256 * 256);
  gn_stats_kernel<<<B_ * G_, 256, 0, stream>>>(x, stats);
  gn_apply_kernel<<<(B_ * N_ * (C_ / 8)) / 256, 256, 0, stream>>>(x, gamma, beta, stats, h);
  qkv_gemm_kernel<<<dim3(N_ / 128, 6, B_), 256, 0, stream>>>(wqkv, h, qkv_b, qb, kb, vb);
  attn_kernel<<<dim3(N_ / 128, B_), 256, 0, stream>>>(qb, kb, vb, ob);
  proj_gemm_kernel<<<dim3(N_ / 128, 2, B_), 256, 0, stream>>>(wproj, ob, proj_b, x, out);
}
